// LeadfieldAdapterCrossAttention_34359738593
// MI455X (gfx1250) — compile-verified, matrix operand order checked by reading, not run on hardware
//
#include <hip/hip_runtime.h>

#define DMODEL 512
#define HDIM   64
#define BB 4
#define NN 8
#define QQ 256
#define KK 1024

typedef __attribute__((ext_vector_type(16))) _Float16 v16h;
typedef __attribute__((ext_vector_type(8)))  _Float16 v8h;
typedef __attribute__((ext_vector_type(8)))  float    v8f;
typedef __attribute__((ext_vector_type(4)))  float    v4f;
typedef __attribute__((ext_vector_type(4)))  unsigned int v4u;
typedef __attribute__((ext_vector_type(8)))  int      v8i;
typedef __attribute__((ext_vector_type(4)))  int      v4i;

__device__ __forceinline__ v8f wmma16x16x32(v16h a, v16h b, v8f c) {
  // D = A(16x32 f16) x B(32x16 f16) + C(16x16 f32)
  return __builtin_amdgcn_wmma_f32_16x16x32_f16(false, a, false, b, (short)0, c,
                                                false, false);
}

// 16-bit operand fragment from row-major memory (global or LDS).
// rowchunk = row base + k-chunk base. Element e of v16h holds
// K = 16*(e>>3) + 8*laneHi + (e&7)  (CDNA5 16-bit A/B VGPR layout)
// => two aligned 16-byte loads per lane.
__device__ __forceinline__ v16h load_frag_h(const _Float16* rowchunk, int laneHi) {
  v8h lo = *(const v8h*)(rowchunk + 8 * laneHi);
  v8h hi = *(const v8h*)(rowchunk + 16 + 8 * laneHi);
  v16h r;
#pragma unroll
  for (int i = 0; i < 8; i++) { r[i] = lo[i]; r[8 + i] = hi[i]; }
  return r;
}

// Same fragment sourced from f32 memory with on-the-fly cvt to f16.
__device__ __forceinline__ v16h load_frag_f32cvt(const float* rowchunk, int laneHi) {
  const float* p = rowchunk + 8 * laneHi;
  v4f a0 = *(const v4f*)(p);
  v4f a1 = *(const v4f*)(p + 4);
  v4f b0 = *(const v4f*)(p + 16);
  v4f b1 = *(const v4f*)(p + 20);
  v16h r;
#pragma unroll
  for (int i = 0; i < 4; i++) {
    r[i]      = (_Float16)a0[i];
    r[4 + i]  = (_Float16)a1[i];
    r[8 + i]  = (_Float16)b0[i];
    r[12 + i] = (_Float16)b1[i];
  }
  return r;
}

#if __has_builtin(__builtin_amdgcn_tensor_load_to_lds) && \
    __has_builtin(__builtin_amdgcn_s_wait_tensorcnt)
#define USE_TDM 1
#else
#define USE_TDM 0
#endif

#if USE_TDM
// Issue one 2-D TDM tile load (tile_h rows x tile_w f16 elements, row stride
// `stride` elements) from global into LDS at byte offset lds_off.
// D# bit layout per CDNA5 ISA ch.8 (groups 2/3 zero => 2-D tensor).
// This toolchain exposes the 6-arg builtin:
//   (uint32x4 g0, int32x8 g1, int32x4 g2, int32x4 g3, int32x8 extra, i32 cpol)
__device__ __forceinline__ void tdm_load_2d(unsigned lds_off,
                                            unsigned long long gaddr,
                                            unsigned tile_w, unsigned tile_h,
                                            unsigned long long stride) {
  v4u g0;
  g0[0] = 1u;                                          // count=1, user descriptor
  g0[1] = lds_off;                                     // lds_addr (bytes)
  g0[2] = (unsigned)(gaddr & 0xffffffffu);             // global_addr[31:0]
  g0[3] = (unsigned)((gaddr >> 32) & 0x1ffffffu)       // global_addr[56:32]
          | (2u << 30);                                // type = 2 ("image")
  v8i g1;
  g1[0] = (int)(1u << 16);                             // wg_mask=0, data_size=1 (2B)
  g1[1] = (int)((tile_w & 0xffffu) << 16);             // tensor_dim0[15:0] (=tile_w)
  g1[2] = (int)(((tile_w >> 16) & 0xffffu)             // tensor_dim0[31:16]
          | ((tile_h & 0xffffu) << 16));               // tensor_dim1[15:0] (=tile_h)
  g1[3] = (int)(((tile_h >> 16) & 0xffffu)             // tensor_dim1[31:16]
          | ((tile_w & 0xffffu) << 16));               // tile_dim0
  g1[4] = (int)(tile_h & 0xffffu);                     // tile_dim1 | tile_dim2=0
  g1[5] = (int)(stride & 0xffffffffu);                 // tensor_dim0_stride[31:0]
  g1[6] = (int)((stride >> 32) & 0xffffu);             // stride[47:32] | d1stride lo=0
  g1[7] = 0;
  v4i z4 = {};
  v8i z8 = {};
  __builtin_amdgcn_tensor_load_to_lds(g0, g1, z4, z4, z8, 0);
}
#endif

// ---------------- weight transpose + f16 convert: Wt[nout][kin] = W[kin][nout]
__global__ void wtrans_kernel(const float* __restrict__ W, _Float16* __restrict__ Wt) {
  const int idx  = blockIdx.x * 256 + threadIdx.x;   // 512*512 threads
  const int nout = idx >> 9;
  const int kin  = idx & 511;
  Wt[idx] = (_Float16)W[kin * DMODEL + nout];
}

// ---------------- projection GEMM: out[M,512] = cvt_f16(A_f32[M,512] @ Wt^T + bias)
__global__ void __launch_bounds__(256)
gemm_f32h(const float* __restrict__ A, const _Float16* __restrict__ Wt,
          const float* __restrict__ bias, _Float16* __restrict__ out,
          int transposeStore) {
  const int lane = threadIdx.x & 31, wave = threadIdx.x >> 5;
  const int laneHi = lane >> 4, laneLo = lane & 15;
  const int m0 = blockIdx.x * 128 + wave * 16;
  const int n0 = blockIdx.y * 128;
  v8f acc[8] = {};
  const float* arow = A + (size_t)(m0 + laneLo) * DMODEL;
  for (int kc = 0; kc < DMODEL; kc += 32) {
    v16h af = load_frag_f32cvt(arow + kc, laneHi);
#pragma unroll
    for (int nt = 0; nt < 8; nt++) {
      const _Float16* wrow = Wt + (size_t)(n0 + nt * 16 + laneLo) * DMODEL + kc;
      v16h bf = load_frag_h(wrow, laneHi);
      acc[nt] = wmma16x16x32(af, bf, acc[nt]);
    }
  }
#pragma unroll
  for (int nt = 0; nt < 8; nt++) {
    const int col = n0 + nt * 16 + laneLo;
    const float bv = bias[col];
#pragma unroll
    for (int r = 0; r < 8; r++) {
      const int row = m0 + r + 8 * laneHi;   // C layout: M = r + 8*laneHi
      const float v = acc[nt][r] + bv;
      size_t idx;
      if (transposeStore)
        idx = ((size_t)((row >> 10) * DMODEL + col)) * KK + (size_t)(row & (KK - 1));
      else
        idx = (size_t)row * DMODEL + col;
      out[idx] = (_Float16)v;
    }
  }
}

// ---------------- final GEMM: out_f32[M,512] = A_f16[M,512] @ WoT^T + bo
__global__ void __launch_bounds__(256)
gemm_h32(const _Float16* __restrict__ A, const _Float16* __restrict__ Wt,
         const float* __restrict__ bias, float* __restrict__ out) {
  const int lane = threadIdx.x & 31, wave = threadIdx.x >> 5;
  const int laneHi = lane >> 4, laneLo = lane & 15;
  const int m0 = blockIdx.x * 128 + wave * 16;
  const int n0 = blockIdx.y * 128;
  v8f acc[8] = {};
  const _Float16* arow = A + (size_t)(m0 + laneLo) * DMODEL;
  for (int kc = 0; kc < DMODEL; kc += 32) {
    v16h af = load_frag_h(arow + kc, laneHi);
#pragma unroll
    for (int nt = 0; nt < 8; nt++) {
      const _Float16* wrow = Wt + (size_t)(n0 + nt * 16 + laneLo) * DMODEL + kc;
      v16h bf = load_frag_h(wrow, laneHi);
      acc[nt] = wmma16x16x32(af, bf, acc[nt]);
    }
  }
#pragma unroll
  for (int nt = 0; nt < 8; nt++) {
    const int col = n0 + nt * 16 + laneLo;
    const float bv = bias[col];
#pragma unroll
    for (int r = 0; r < 8; r++) {
      const int row = m0 + r + 8 * laneHi;
      out[(size_t)row * DMODEL + col] = acc[nt][r] + bv;
    }
  }
}

// ---------------- positional bias: bias[b][q][k] = -200*dist2 + lf/sqrt(8)
__global__ void bias_kernel(const float* __restrict__ qpos, const float* __restrict__ kpos,
                            const float* __restrict__ Wqb, const float* __restrict__ Wkb,
                            float* __restrict__ biasOut) {
  const int idx = blockIdx.x * 256 + threadIdx.x;   // B*Q*K = 1048576 threads
  const int k = idx & (KK - 1);
  const int q = (idx >> 10) & (QQ - 1);
  const int b = idx >> 18;
  const float* qp = qpos + ((size_t)b * QQ + q) * 3;
  const float* kp = kpos + ((size_t)b * KK + k) * 3;
  const float dx = qp[0] - kp[0], dy = qp[1] - kp[1], dz = qp[2] - kp[2];
  const float dist2 = dx * dx + dy * dy + dz * dz;
  float lf = 0.f;
#pragma unroll
  for (int r = 0; r < 8; r++) {
    const float qb = qp[0] * Wqb[r] + qp[1] * Wqb[8 + r] + qp[2] * Wqb[16 + r];
    const float kb = kp[0] * Wkb[r] + kp[1] * Wkb[8 + r] + kp[2] * Wkb[16 + r];
    lf += qb * kb;
  }
  biasOut[idx] = -dist2 * 200.0f + lf * 0.35355339059327373f;
}

// ---------------- flash attention per (b,n,h), TDM-staged K/V, double buffered
// grid = 512 blocks: [qblk(2)] x [bnh(256)] ; 8 waves, wave = 16 query rows.
__global__ void __launch_bounds__(256)
attn_kernel(const _Float16* __restrict__ qh, const _Float16* __restrict__ kh,
            const _Float16* __restrict__ vT, const float* __restrict__ biasArr,
            const unsigned char* __restrict__ mask, _Float16* __restrict__ ctx) {
  __shared__ __align__(16) _Float16 Ktile[2][64 * 64];   // [key][d]   8 KB x2
  __shared__ __align__(16) _Float16 Vtile[2][64 * 64];   // [d][key]   8 KB x2
  __shared__ __align__(16) _Float16 Pbuf[8 * 16 * 64];   // per-wave P  16 KB
  const int lane = threadIdx.x & 31, wave = threadIdx.x >> 5;
  const int laneHi = lane >> 4, laneLo = lane & 15;
  const int blk  = blockIdx.x;
  const int qblk = blk & 1;
  const int bnh  = blk >> 1;
  const int h  = bnh & 7;
  const int bn = bnh >> 3;       // b*8 + n
  const int b  = bn >> 3;
  const int q0 = qblk * 128 + wave * 16;
  const float scale = 0.125f;    // 64^-0.5

  const size_t krowbase = (size_t)bn * KK;
  const _Float16* kbase = kh + krowbase * DMODEL + h * HDIM;          // + key*512
  const _Float16* vbase = vT + ((size_t)(bn * DMODEL + h * HDIM)) * KK; // + d*1024

  // stage key/value tile `kt` into LDS buffer `buf`
  auto stage = [&](int kt, int buf) {
#if USE_TDM
    if (wave == 0) {
      const unsigned koff = (unsigned)(unsigned long long)(const void*)&Ktile[buf][0];
      const unsigned voff = (unsigned)(unsigned long long)(const void*)&Vtile[buf][0];
      tdm_load_2d(koff, (unsigned long long)(const void*)(kbase + (size_t)kt * 64 * DMODEL),
                  64, 64, DMODEL);
      tdm_load_2d(voff, (unsigned long long)(const void*)(vbase + kt * 64),
                  64, 64, KK);
    }
#else
    // cooperative fallback: 256 threads copy 2x4KB (8 halves each per tile, rows 0-31)
    const int t = threadIdx.x;
    const int row = t >> 3, seg = t & 7;               // 8 x 8-half segments per row
    const v8h* ks = (const v8h*)(kbase + (size_t)(kt * 64 + row) * DMODEL) + seg;
    ((v8h*)&Ktile[buf][row * 64])[seg] = *ks;
    const v8h* vs = (const v8h*)(vbase + (size_t)row * KK + kt * 64) + seg;
    ((v8h*)&Vtile[buf][row * 64])[seg] = *vs;
#endif
  };

  // q fragment for this wave's 16 rows (d = 64 -> two 32-K chunks), loaded once
  const _Float16* qrow = qh + ((size_t)(bn * QQ + q0 + laneLo)) * DMODEL + h * HDIM;
  const v16h qa0 = load_frag_h(qrow, laneHi);
  const v16h qa1 = load_frag_h(qrow + 32, laneHi);

  v8f cacc[4] = {};              // 16x64 context accumulator (4 d-tiles)
  float mrow[8], lrow[8];
#pragma unroll
  for (int r = 0; r < 8; r++) { mrow[r] = -__builtin_inff(); lrow[r] = 0.f; }

  _Float16* Pw = &Pbuf[wave * 16 * 64];
  const float* bias_b = biasArr + (size_t)b * QQ * KK;

  // prologue: land tile 0
  stage(0, 0);
#if USE_TDM
  if (wave == 0) __builtin_amdgcn_s_wait_tensorcnt(0);
#endif
  __syncthreads();

  for (int kt = 0; kt < 16; kt++) {
    const int cur = kt & 1;
    if (kt < 15) stage(kt + 1, cur ^ 1);   // DMA next tile under current compute
    const int k0 = kt * 64;

    // ---- S = q @ k^T from LDS K tile (4 key-tiles x 2 K-chunks = 8 WMMA)
    v8f sArr[4] = {};
#pragma unroll
    for (int nt = 0; nt < 4; nt++) {
      const _Float16* krow = &Ktile[cur][(nt * 16 + laneLo) * 64];
      v16h kb0 = load_frag_h(krow, laneHi);
      sArr[nt] = wmma16x16x32(qa0, kb0, sArr[nt]);
      v16h kb1 = load_frag_h(krow + 32, laneHi);
      sArr[nt] = wmma16x16x32(qa1, kb1, sArr[nt]);
    }
    // ---- scale + bias + mask (C layout: M=r+8*laneHi, N=laneLo)
    float logit[4][8];
#pragma unroll
    for (int nt = 0; nt < 4; nt++) {
      const int kcol = k0 + nt * 16 + laneLo;
      const bool valid = mask[b * KK + kcol] != 0;
      const float* bp = bias_b + kcol;
#pragma unroll
      for (int r = 0; r < 8; r++) {
        const int qg = q0 + r + 8 * laneHi;
        const float l = sArr[nt][r] * scale + bp[(size_t)qg * KK];
        logit[nt][r] = valid ? l : -10000.0f;
      }
    }
    // ---- online softmax (row lives in one 16-lane half -> xor-shuffle reduce)
#pragma unroll
    for (int r = 0; r < 8; r++) {
      float tmax = fmaxf(fmaxf(logit[0][r], logit[1][r]),
                         fmaxf(logit[2][r], logit[3][r]));
#pragma unroll
      for (int mk = 1; mk < 16; mk <<= 1) tmax = fmaxf(tmax, __shfl_xor(tmax, mk, 32));
      const float mnew = fmaxf(mrow[r], tmax);
      const float corr = __expf(mrow[r] - mnew);
      mrow[r] = mnew;
      float ps = 0.f;
#pragma unroll
      for (int nt = 0; nt < 4; nt++) {
        const float p = __expf(logit[nt][r] - mnew);
        logit[nt][r] = p;
        ps += p;
      }
#pragma unroll
      for (int mk = 1; mk < 16; mk <<= 1) ps += __shfl_xor(ps, mk, 32);
      lrow[r] = lrow[r] * corr + ps;
#pragma unroll
      for (int dt = 0; dt < 4; dt++) cacc[dt][r] *= corr;
    }
    // ---- P (C layout) -> LDS row-major 16x64, reload in A-fragment layout
#pragma unroll
    for (int nt = 0; nt < 4; nt++)
#pragma unroll
      for (int r = 0; r < 8; r++)
        Pw[(r + 8 * laneHi) * 64 + nt * 16 + laneLo] = (_Float16)logit[nt][r];
    asm volatile("s_wait_dscnt 0" ::: "memory");   // wave-private strip, DS in-order
    const v16h pa0 = load_frag_h(Pw + laneLo * 64, laneHi);
    const v16h pa1 = load_frag_h(Pw + laneLo * 64 + 32, laneHi);
    // ---- ctx += P @ V from LDS V tile (4 d-tiles x 2 K-chunks = 8 WMMA)
#pragma unroll
    for (int dt = 0; dt < 4; dt++) {
      const _Float16* vrow = &Vtile[cur][(dt * 16 + laneLo) * 64];
      v16h vb0 = load_frag_h(vrow, laneHi);
      cacc[dt] = wmma16x16x32(pa0, vb0, cacc[dt]);
      v16h vb1 = load_frag_h(vrow + 32, laneHi);
      cacc[dt] = wmma16x16x32(pa1, vb1, cacc[dt]);
    }
    __syncthreads();                     // everyone done reading buffer `cur`
#if USE_TDM
    if (wave == 0 && kt < 15) __builtin_amdgcn_s_wait_tensorcnt(0);
#endif
    __syncthreads();                     // next buffer fully landed
  }
  // ---- normalize and emit ctx rows (f16, row-major [bn][q][h*64+d])
#pragma unroll
  for (int r = 0; r < 8; r++) {
    const float inv = 1.0f / lrow[r];
    const int qg = q0 + r + 8 * laneHi;
    _Float16* orow = ctx + ((size_t)(bn * QQ + qg)) * DMODEL + h * HDIM + laneLo;
#pragma unroll
    for (int dt = 0; dt < 4; dt++)
      orow[dt * 16] = (_Float16)(cacc[dt][r] * inv);
  }
}

extern "C" void kernel_launch(void* const* d_in, const int* in_sizes, int n_in,
                              void* d_out, int out_size, void* d_ws, size_t ws_size,
                              hipStream_t stream) {
  (void)in_sizes; (void)n_in; (void)out_size; (void)ws_size;
  const float* query     = (const float*)d_in[0];
  const float* key_value = (const float*)d_in[1];
  const float* query_pos = (const float*)d_in[2];
  const float* key_pos   = (const float*)d_in[3];
  const unsigned char* key_mask = (const unsigned char*)d_in[4];
  const float* Wq = (const float*)d_in[5];
  const float* bq = (const float*)d_in[6];
  const float* Wk = (const float*)d_in[7];
  const float* bk = (const float*)d_in[8];
  const float* Wv = (const float*)d_in[9];
  const float* bv = (const float*)d_in[10];
  const float* Wo = (const float*)d_in[11];
  const float* bo = (const float*)d_in[12];
  const float* Wqb = (const float*)d_in[13];
  const float* Wkb = (const float*)d_in[14];

  char* ws = (char*)d_ws;
  _Float16* WqT = (_Float16*)ws; ws += (size_t)DMODEL * DMODEL * 2;
  _Float16* WkT = (_Float16*)ws; ws += (size_t)DMODEL * DMODEL * 2;
  _Float16* WvT = (_Float16*)ws; ws += (size_t)DMODEL * DMODEL * 2;
  _Float16* WoT = (_Float16*)ws; ws += (size_t)DMODEL * DMODEL * 2;
  _Float16* qh  = (_Float16*)ws; ws += (size_t)BB * NN * QQ * DMODEL * 2;  // 8 MB
  _Float16* khb = (_Float16*)ws; ws += (size_t)BB * NN * KK * DMODEL * 2;  // 32 MB
  _Float16* vTb = (_Float16*)ws; ws += (size_t)BB * NN * KK * DMODEL * 2;  // 32 MB
  float*    bia = (float*)ws;    ws += (size_t)BB * QQ * KK * 4;           // 4 MB
  _Float16* ctx = (_Float16*)ws; ws += (size_t)BB * NN * QQ * DMODEL * 2;  // 8 MB

  wtrans_kernel<<<1024, 256, 0, stream>>>(Wq, WqT);
  wtrans_kernel<<<1024, 256, 0, stream>>>(Wk, WkT);
  wtrans_kernel<<<1024, 256, 0, stream>>>(Wv, WvT);
  wtrans_kernel<<<1024, 256, 0, stream>>>(Wo, WoT);

  gemm_f32h<<<dim3(64, 4), 256, 0, stream>>>(query, WqT, bq, qh, 0);
  gemm_f32h<<<dim3(256, 4), 256, 0, stream>>>(key_value, WkT, bk, khb, 0);
  gemm_f32h<<<dim3(256, 4), 256, 0, stream>>>(key_value, WvT, bv, vTb, 1);

  bias_kernel<<<4096, 256, 0, stream>>>(query_pos, key_pos, Wqb, Wkb, bia);

  attn_kernel<<<512, 256, 0, stream>>>(qh, khb, vTb, bia, key_mask, ctx);

  gemm_h32<<<dim3(64, 4), 256, 0, stream>>>(ctx, WoT, bo, (float*)d_out);
}
